// GNNEval_2413771620643
// MI455X (gfx1250) — matrix operand order repaired and checked by reading, not compiled, not run on hardware
//
#include <hip/hip_runtime.h>
#include <hip/hip_bf16.h>

// ---------- types for WMMA ----------
typedef __bf16 bf16_t;
typedef __attribute__((ext_vector_type(16))) __bf16 v16bf;
typedef __attribute__((ext_vector_type(8)))  __bf16 v8bf;
typedef __attribute__((ext_vector_type(8)))  float  v8f;

#define N_NODES   100000
#define N_EDGES   300000
#define NUM_GRAPHS 64
#define HID       256
#define BN_INV    0.99999500003749967f   // 1/sqrt(1+1e-5)

__device__ __forceinline__ bf16_t f2bf(float f) {
    union { float f; unsigned u; } x; x.f = f;
    unsigned short s = (unsigned short)((x.u + 0x7FFFu + ((x.u >> 16) & 1u)) >> 16);
    return __builtin_bit_cast(bf16_t, s);
}

// ---------- small utility kernels ----------
__global__ void k_copy(const float* __restrict__ src, float* __restrict__ dst, long n) {
    long i = (long)blockIdx.x * blockDim.x + threadIdx.x;
    if (i < n) dst[i] = src[i];
}

__global__ void k_zero(float* __restrict__ dst, long n) {
    long i = (long)blockIdx.x * blockDim.x + threadIdx.x;
    if (i < n) dst[i] = 0.0f;
}

// W [K][256] f32 row-major  ->  Wt [256][K] bf16 (column-major of original)
__global__ void k_wt(const float* __restrict__ W, bf16_t* __restrict__ Wt, int K) {
    int gid = blockIdx.x * blockDim.x + threadIdx.x;
    int total = K * HID;
    if (gid >= total) return;
    int k = gid / HID, c = gid % HID;
    Wt[c * K + k] = f2bf(W[gid]);
}

// ---------- edge message + scatter-add ----------
// msg = relu(h[src] + edge_attr @ leW + leb); Z[dst] += msg   (Z pre-loaded with h)
__global__ void k_edge_msg(const int* __restrict__ ei, const float* __restrict__ ea,
                           const float* __restrict__ leW, const float* __restrict__ leb,
                           const float* __restrict__ h, float* __restrict__ Z,
                           int dshift) {
    long gid = (long)blockIdx.x * blockDim.x + threadIdx.x;
    int  d   = 1 << dshift;
    long total = (long)N_EDGES << dshift;
    if (gid >= total) return;
    int e = (int)(gid >> dshift);
    int j = (int)(gid & (d - 1));
    int s  = ei[e];
    int dn = ei[N_EDGES + e];
    float a0 = ea[2 * e], a1 = ea[2 * e + 1];
    float ev = a0 * leW[j] + a1 * leW[d + j] + leb[j];
    float m  = h[(long)s * d + j] + ev;
    m = m > 0.0f ? m : 0.0f;
    unsafeAtomicAdd(&Z[(long)dn * d + j], m);
}

// ---------- fused 2-GEMM node MLP (WMMA bf16 -> f32) ----------
// Out[node] = relu(BN(relu(Z[node]@W1+b1)@W2+b2)) (+resid).  16 nodes/block, 8 waves.
__global__ __launch_bounds__(256) void k_fused_mlp(
    const float* __restrict__ Zin, int K1,
    const bf16_t* __restrict__ W1t, const float* __restrict__ b1,
    const bf16_t* __restrict__ W2t, const float* __restrict__ b2,
    const float* __restrict__ gamma, const float* __restrict__ beta,
    const float* __restrict__ resid,      // nullable
    float* __restrict__ Out) {

    __shared__ __align__(16) bf16_t zt[16 * (HID + 8)];   // z tile, stride K1+8
    __shared__ __align__(16) bf16_t tt[16 * (HID + 8)];   // mid tile, stride 264

    const int tid   = threadIdx.x;
    const int node0 = blockIdx.x * 16;
    const int sA    = K1 + 8;
    const int sT    = HID + 8;

    // stage z tile (f32 -> bf16) into LDS
    for (int i = tid; i < 16 * K1; i += 256) {
        int m = i / K1, k = i - m * K1;
        zt[m * sA + k] = f2bf(Zin[(long)(node0 + m) * K1 + k]);
    }
    __syncthreads();

    const int w    = tid >> 5;
    const int lane = tid & 31;
    const int n    = lane & 15;     // column within 16-wide tile / A row M
    const int hi   = lane >> 4;     // K-half selector (and C-row-half selector)
    const int c0   = w * 32, c1 = c0 + 16;

    // ---- GEMM A: t = relu(z @ W1 + b1) ----
    v8f acc0 = {}, acc1 = {};
    for (int kb = 0; kb < K1; kb += 32) {
        const bf16_t* ar = zt + n * sA + kb;
        v8bf alo = *(const v8bf*)(ar + hi * 8);
        v8bf ahi = *(const v8bf*)(ar + 16 + hi * 8);
        v16bf a;
        #pragma unroll
        for (int i = 0; i < 8; ++i) { a[i] = alo[i]; a[8 + i] = ahi[i]; }

        const bf16_t* bp0 = W1t + (long)(c0 + n) * K1 + kb + hi * 8;
        v8bf b0l = *(const v8bf*)(bp0);
        v8bf b0h = *(const v8bf*)(bp0 + 16);
        v16bf b0;
        #pragma unroll
        for (int i = 0; i < 8; ++i) { b0[i] = b0l[i]; b0[8 + i] = b0h[i]; }
        acc0 = __builtin_amdgcn_wmma_f32_16x16x32_bf16(false, a, false, b0,
                                                       (short)0, acc0, false, false);

        const bf16_t* bp1 = W1t + (long)(c1 + n) * K1 + kb + hi * 8;
        v8bf b1l = *(const v8bf*)(bp1);
        v8bf b1h = *(const v8bf*)(bp1 + 16);
        v16bf bv1;
        #pragma unroll
        for (int i = 0; i < 8; ++i) { bv1[i] = b1l[i]; bv1[8 + i] = b1h[i]; }
        acc1 = __builtin_amdgcn_wmma_f32_16x16x32_bf16(false, a, false, bv1,
                                                       (short)0, acc1, false, false);
    }
    {
        float bias0 = b1[c0 + n], bias1 = b1[c1 + n];
        #pragma unroll
        for (int r = 0; r < 8; ++r) {
            int m = hi * 8 + r;
            float t0 = acc0[r] + bias0; t0 = t0 > 0.0f ? t0 : 0.0f;
            float t1 = acc1[r] + bias1; t1 = t1 > 0.0f ? t1 : 0.0f;
            tt[m * sT + c0 + n] = f2bf(t0);
            tt[m * sT + c1 + n] = f2bf(t1);
        }
    }
    __syncthreads();

    // ---- GEMM B: out = t @ W2 + b2 ; BN affine ; relu ; (+resid) ----
    v8f accb0 = {}, accb1 = {};
    for (int kb = 0; kb < HID; kb += 32) {
        const bf16_t* ar = tt + n * sT + kb;
        v8bf alo = *(const v8bf*)(ar + hi * 8);
        v8bf ahi = *(const v8bf*)(ar + 16 + hi * 8);
        v16bf a;
        #pragma unroll
        for (int i = 0; i < 8; ++i) { a[i] = alo[i]; a[8 + i] = ahi[i]; }

        const bf16_t* bp0 = W2t + (long)(c0 + n) * HID + kb + hi * 8;
        v8bf b0l = *(const v8bf*)(bp0);
        v8bf b0h = *(const v8bf*)(bp0 + 16);
        v16bf b0;
        #pragma unroll
        for (int i = 0; i < 8; ++i) { b0[i] = b0l[i]; b0[8 + i] = b0h[i]; }
        accb0 = __builtin_amdgcn_wmma_f32_16x16x32_bf16(false, a, false, b0,
                                                        (short)0, accb0, false, false);

        const bf16_t* bp1 = W2t + (long)(c1 + n) * HID + kb + hi * 8;
        v8bf b1l = *(const v8bf*)(bp1);
        v8bf b1h = *(const v8bf*)(bp1 + 16);
        v16bf bv1;
        #pragma unroll
        for (int i = 0; i < 8; ++i) { bv1[i] = b1l[i]; bv1[8 + i] = b1h[i]; }
        accb1 = __builtin_amdgcn_wmma_f32_16x16x32_bf16(false, a, false, bv1,
                                                        (short)0, accb1, false, false);
    }
    {
        float bb0 = b2[c0 + n], sc0 = gamma[c0 + n] * BN_INV, be0 = beta[c0 + n];
        float bb1 = b2[c1 + n], sc1 = gamma[c1 + n] * BN_INV, be1 = beta[c1 + n];
        #pragma unroll
        for (int r = 0; r < 8; ++r) {
            int m = hi * 8 + r;
            long node = node0 + m;
            float v0 = (accb0[r] + bb0) * sc0 + be0; v0 = v0 > 0.0f ? v0 : 0.0f;
            float v1 = (accb1[r] + bb1) * sc1 + be1; v1 = v1 > 0.0f ? v1 : 0.0f;
            if (resid) {
                v0 += resid[node * HID + c0 + n];
                v1 += resid[node * HID + c1 + n];
            }
            Out[node * HID + c0 + n] = v0;
            Out[node * HID + c1 + n] = v1;
        }
    }
}

// ---------- mean pooling (atomics) ----------
__global__ void k_pool(const float* __restrict__ h, const int* __restrict__ batch,
                       float* __restrict__ sums, float* __restrict__ counts) {
    long gid = (long)blockIdx.x * blockDim.x + threadIdx.x;
    long total = (long)N_NODES * HID;
    if (gid >= total) return;
    int node = (int)(gid >> 8);
    int j    = (int)(gid & (HID - 1));
    int g = batch[node];
    unsafeAtomicAdd(&sums[g * HID + j], h[gid]);
    if (j == 0) unsafeAtomicAdd(&counts[g], 1.0f);
}

// ---------- head: relu(concat(pooled, g) @ Wh1 + bh1) @ Wh2 + bh2 ----------
__global__ __launch_bounds__(256) void k_head(
    const float* __restrict__ sums, const float* __restrict__ counts,
    const float* __restrict__ gfeat,
    const float* __restrict__ Wh1, const float* __restrict__ bh1,
    const float* __restrict__ Wh2, const float* __restrict__ bh2,
    float* __restrict__ out) {
    __shared__ float feat[HID + 16];
    __shared__ float red[HID];
    int g = blockIdx.x, tid = threadIdx.x;
    float cnt = counts[g]; cnt = cnt > 1.0f ? cnt : 1.0f;
    feat[tid] = sums[g * HID + tid] / cnt;
    if (tid < 16) feat[HID + tid] = gfeat[g * 16 + tid];
    __syncthreads();
    float acc = bh1[tid];
    for (int k = 0; k < HID + 16; ++k) acc += feat[k] * Wh1[k * HID + tid];
    acc = acc > 0.0f ? acc : 0.0f;
    red[tid] = acc * Wh2[tid];
    __syncthreads();
    for (int s = 128; s > 0; s >>= 1) {
        if (tid < s) red[tid] += red[tid + s];
        __syncthreads();
    }
    if (tid == 0) out[g] = red[0] + bh2[0];
}

// ---------- driver ----------
extern "C" void kernel_launch(void* const* d_in, const int* in_sizes, int n_in,
                              void* d_out, int out_size, void* d_ws, size_t ws_size,
                              hipStream_t stream) {
    (void)in_sizes; (void)n_in; (void)out_size; (void)ws_size;

    const float* x  = (const float*)d_in[0];
    const int*   ei = (const int*)  d_in[1];
    const float* ea = (const float*)d_in[2];
    const int*   bt = (const int*)  d_in[3];
    const float* gg = (const float*)d_in[4];

    // per-layer params start at index 5, 8 entries each: leW leb W1 b1 W2 b2 gamma beta
    const float* leW[3]; const float* leb[3];
    const float* W1[3];  const float* b1[3];
    const float* W2[3];  const float* b2[3];
    const float* gam[3]; const float* bet[3];
    for (int l = 0; l < 3; ++l) {
        int b = 5 + 8 * l;
        leW[l] = (const float*)d_in[b + 0];
        leb[l] = (const float*)d_in[b + 1];
        W1[l]  = (const float*)d_in[b + 2];
        b1[l]  = (const float*)d_in[b + 3];
        W2[l]  = (const float*)d_in[b + 4];
        b2[l]  = (const float*)d_in[b + 5];
        gam[l] = (const float*)d_in[b + 6];
        bet[l] = (const float*)d_in[b + 7];
    }
    const float* Wh1 = (const float*)d_in[29];
    const float* bh1 = (const float*)d_in[30];
    const float* Wh2 = (const float*)d_in[31];
    const float* bh2 = (const float*)d_in[32];

    // workspace carve (256B aligned)
    char* p = (char*)d_ws;
    auto carve = [&](size_t bytes) -> void* {
        void* r = (void*)p; p += (bytes + 255) & ~(size_t)255; return r;
    };
    float*  bufA  = (float*)carve((size_t)N_NODES * HID * 4);
    float*  bufB  = (float*)carve((size_t)N_NODES * HID * 4);
    bf16_t* W1t[3]; bf16_t* W2t[3];
    for (int l = 0; l < 3; ++l) {
        W1t[l] = (bf16_t*)carve((size_t)HID * HID * 2);
        W2t[l] = (bf16_t*)carve((size_t)HID * HID * 2);
    }
    float* sums   = (float*)carve((size_t)NUM_GRAPHS * HID * 4);
    float* counts = (float*)carve((size_t)NUM_GRAPHS * 4);

    const int Kin[3] = {32, HID, HID};

    // 1) convert + transpose weights to bf16 column-major
    for (int l = 0; l < 3; ++l) {
        int t1 = Kin[l] * HID;
        k_wt<<<(t1 + 255) / 256, 256, 0, stream>>>(W1[l], W1t[l], Kin[l]);
        k_wt<<<(HID * HID + 255) / 256, 256, 0, stream>>>(W2[l], W2t[l], HID);
    }

    const int mlpBlocks = N_NODES / 16;   // 6250

    // ---- layer 0: h=x (d=32), aggr->bufA, mlp->bufB ----
    {
        long n = (long)N_NODES * 32;
        k_copy<<<(int)((n + 255) / 256), 256, 0, stream>>>(x, bufA, n);
        long te = (long)N_EDGES << 5;
        k_edge_msg<<<(int)((te + 255) / 256), 256, 0, stream>>>(ei, ea, leW[0], leb[0], x, bufA, 5);
        k_fused_mlp<<<mlpBlocks, 256, 0, stream>>>(bufA, 32, W1t[0], b1[0], W2t[0], b2[0],
                                                   gam[0], bet[0], nullptr, bufB);
    }
    // ---- layer 1: h=bufB, aggr->bufA, mlp in-place bufA, resid=bufB ----
    {
        long n = (long)N_NODES * HID;
        k_copy<<<(int)((n + 255) / 256), 256, 0, stream>>>(bufB, bufA, n);
        long te = (long)N_EDGES << 8;
        k_edge_msg<<<(int)((te + 255) / 256), 256, 0, stream>>>(ei, ea, leW[1], leb[1], bufB, bufA, 8);
        k_fused_mlp<<<mlpBlocks, 256, 0, stream>>>(bufA, HID, W1t[1], b1[1], W2t[1], b2[1],
                                                   gam[1], bet[1], bufB, bufA);
    }
    // ---- layer 2: h=bufA, aggr->bufB, mlp in-place bufB, resid=bufA ----
    {
        long n = (long)N_NODES * HID;
        k_copy<<<(int)((n + 255) / 256), 256, 0, stream>>>(bufA, bufB, n);
        long te = (long)N_EDGES << 8;
        k_edge_msg<<<(int)((te + 255) / 256), 256, 0, stream>>>(ei, ea, leW[2], leb[2], bufA, bufB, 8);
        k_fused_mlp<<<mlpBlocks, 256, 0, stream>>>(bufB, HID, W1t[2], b1[2], W2t[2], b2[2],
                                                   gam[2], bet[2], bufA, bufB);
    }
    // ---- pooling + head ----
    k_zero<<<(NUM_GRAPHS * HID + 255) / 256, 256, 0, stream>>>(sums, NUM_GRAPHS * HID);
    k_zero<<<1, 256, 0, stream>>>(counts, NUM_GRAPHS);
    {
        long n = (long)N_NODES * HID;
        k_pool<<<(int)((n + 255) / 256), 256, 0, stream>>>(bufB, bt, sums, counts);
    }
    k_head<<<NUM_GRAPHS, 256, 0, stream>>>(sums, counts, gg, Wh1, bh1, Wh2, bh2, (float*)d_out);
}
